// ImprovedEnergyHead_33792802685620
// MI455X (gfx1250) — matrix operand order repaired and checked by reading, not compiled, not run on hardware
//
#include <hip/hip_runtime.h>

#define DEV_INLINE __device__ __forceinline__

typedef __attribute__((ext_vector_type(16))) __bf16 v16bf;
typedef __attribute__((ext_vector_type(8)))  float  v8f;
typedef __attribute__((ext_vector_type(2)))  float  v2f;

struct u4 { unsigned int x, y, z, w; };
union Frag16 { v16bf v; u4 q[2]; };

static constexpr int Bn  = 4;
static constexpr int Nn  = 2048;
static constexpr int Dn  = 1024;
static constexpr int DFF = 4096;
static constexpr int DP  = 256;

// workspace layout (bytes)
static constexpr size_t OFF_G     = 0;       // B*D floats (mean-pooled g)
static constexpr size_t OFF_S     = 16384;   // B*DP floats (sum of xp over tokens)
static constexpr size_t OFF_LOCAL = 20480;   // B floats   (local energy sums)
static constexpr size_t OFF_GE    = 20496;   // B floats   (global energy sums)
static constexpr size_t OFF_SSQ   = 20512;   // B doubles  (sum ||xp_n||^2)
static constexpr int    ZERO_DW   = (int)((20544 - OFF_S) / 4);

DEV_INLINE unsigned short f2bf(float f) {
  unsigned int u = __float_as_uint(f);
  u += 0x7FFFu + ((u >> 16) & 1u);          // round-to-nearest-even
  return (unsigned short)(u >> 16);
}

DEV_INLINE float gelu_exact(float t) {
  return 0.5f * t * (1.0f + erff(t * 0.70710678118654752f));
}

// ---- CDNA5 async LDS-DMA (ASYNCcnt path) via inline asm ----
// GLOBAL_LOAD_ASYNC_TO_LDS_B128: vdst = LDS byte offset (low 32 bits of generic
// LDS pointer; aperture base has zero low bits), vaddr = 64-bit global address.
DEV_INLINE void async_b128(const float* g, float* l) {
  unsigned ldsoff = (unsigned)(size_t)l;
  unsigned long long ga = (unsigned long long)(size_t)g;
  asm volatile("global_load_async_to_lds_b128 %0, %1, off"
               :: "v"(ldsoff), "v"(ga)
               : "memory");
}
DEV_INLINE void async_copy32B(const float* g, float* l) {  // 8 floats
  async_b128(g, l);
  async_b128(g + 4, l + 4);
}
DEV_INLINE void wait_async0() {
#if __has_builtin(__builtin_amdgcn_s_wait_asynccnt)
  __builtin_amdgcn_s_wait_asynccnt(0);
#else
  asm volatile("s_wait_asynccnt 0x0" ::: "memory");
#endif
}

// ---------------- zero accumulators (graph-capture safe) ----------------
__global__ void k_zero(unsigned int* p, int n) {
  int i = blockIdx.x * blockDim.x + threadIdx.x;
  if (i < n) p[i] = 0u;
}

// ---------------- g[b,d] = mean_n x[b,n,d] ----------------
__global__ __launch_bounds__(256) void k_colmean(const float* __restrict__ x,
                                                 float* __restrict__ g) {
  int i = blockIdx.x * blockDim.x + threadIdx.x;   // 0..B*D-1
  int b = i >> 10, d = i & 1023;
  const float* p = x + (size_t)b * Nn * Dn + d;
  float s = 0.f;
  for (int n = 0; n < Nn; ++n) {
    __builtin_prefetch(p + (size_t)(n + 32) * Dn, 0, 0);  // speculative ok
    s += p[(size_t)n * Dn];
  }
  g[i] = s * (1.0f / Nn);
}

// ---------------- fused local-energy MLP: bf16 WMMA 16x16x32 ----------------
// grid (64, 32): block tile 128 rows (of B*N) x 128 cols (of DFF)
__global__ __launch_bounds__(256) void k_localmlp(const float* __restrict__ x,
                                                  const float* __restrict__ lw1,
                                                  const float* __restrict__ lb1,
                                                  const float* __restrict__ lw2,
                                                  float* __restrict__ localAcc) {
  __shared__ __align__(16) unsigned short As[128 * 40];  // [m][k] bf16, stride 40
  __shared__ __align__(16) unsigned short Bs[128 * 40];  // [n][k] bf16 (transposed)
  __shared__ float red[256];

  const int tid  = threadIdx.x;
  const int lane = tid & 31, w = tid >> 5;
  const int wm = w & 3, wn = w >> 2;       // 4x2 wave grid -> each wave 32x64
  const int l15 = lane & 15, lh = lane >> 4;
  const int bm = blockIdx.x, bn = blockIdx.y;
  const size_t rowBase = (size_t)bm * 128;
  const int colBase = bn * 128;

  const v8f vzero = {0.f, 0.f, 0.f, 0.f, 0.f, 0.f, 0.f, 0.f};
  v8f acc[2][4];
#pragma unroll
  for (int mi = 0; mi < 2; ++mi)
#pragma unroll
    for (int ni = 0; ni < 4; ++ni) acc[mi][ni] = vzero;

  for (int k0 = 0; k0 < Dn; k0 += 32) {
    // stage A (x -> bf16): 128x32 (convert once per element at store time)
    {
      int r = tid >> 1, half = tid & 1;
      const float* src = x + (rowBase + r) * (size_t)Dn + k0 + half * 16;
      unsigned short* dst = &As[r * 40 + half * 16];
#pragma unroll
      for (int j = 0; j < 4; ++j) {
        float4 v = ((const float4*)src)[j];
        dst[j * 4 + 0] = f2bf(v.x);
        dst[j * 4 + 1] = f2bf(v.y);
        dst[j * 4 + 2] = f2bf(v.z);
        dst[j * 4 + 3] = f2bf(v.w);
      }
    }
    // stage B (lw1 -> bf16, transposed to [n][k]): 32x128
    {
      int c = tid & 127, kk0 = (tid >> 7) * 16;
      const float* src = lw1 + (size_t)(k0 + kk0) * DFF + colBase + c;
      unsigned short* dst = &Bs[c * 40 + kk0];
#pragma unroll
      for (int j = 0; j < 16; ++j) dst[j] = f2bf(src[(size_t)j * DFF]);
    }
    __syncthreads();

    Frag16 af[2], bfg[4];
#pragma unroll
    for (int mi = 0; mi < 2; ++mi) {
      int ro = (wm * 32 + mi * 16 + l15) * 40 + lh * 8;   // A: k = 8*lh(+16)
      af[mi].q[0] = *(const u4*)&As[ro];
      af[mi].q[1] = *(const u4*)&As[ro + 16];
    }
#pragma unroll
    for (int ni = 0; ni < 4; ++ni) {
      int bo = (wn * 64 + ni * 16 + l15) * 40 + lh * 8;   // B: k = 8*lh(+16)
      bfg[ni].q[0] = *(const u4*)&Bs[bo];
      bfg[ni].q[1] = *(const u4*)&Bs[bo + 16];
    }
#pragma unroll
    for (int mi = 0; mi < 2; ++mi)
#pragma unroll
      for (int ni = 0; ni < 4; ++ni)
        acc[mi][ni] = __builtin_amdgcn_wmma_f32_16x16x32_bf16(
            false, af[mi].v, false, bfg[ni].v, (short)0, acc[mi][ni], false, false);
    __syncthreads();
  }

  // epilogue: gelu(acc+lb1) * lw2, reduce to per-batch scalar
  float lsum = 0.f;
#pragma unroll
  for (int ni = 0; ni < 4; ++ni) {
    int col = colBase + wn * 64 + ni * 16 + l15;
    float bias = lb1[col];
    float w2   = lw2[col];
#pragma unroll
    for (int mi = 0; mi < 2; ++mi)
#pragma unroll
      for (int v = 0; v < 8; ++v) {
        float t = acc[mi][ni][v] + bias;
        lsum += gelu_exact(t) * w2;
      }
  }
  red[tid] = lsum;
  __syncthreads();
  for (int s = 128; s > 0; s >>= 1) {
    if (tid < s) red[tid] += red[tid + s];
    __syncthreads();
  }
  if (tid == 0) atomicAdd(&localAcc[bm >> 4], red[0]);  // 16 row-blocks per batch
}

// ---------------- interaction: xp = x@pw + pb, fp32 WMMA 16x16x4 ----------------
// Double-buffered async-LDS pipeline. grid (64, 2): 128 rows x 128 cols of DP
static constexpr int ASTR = 20;   // A tile [m][k] stride (floats)
static constexpr int BSTR = 136;  // B tile [k][n] stride (floats)

__global__ __launch_bounds__(256) void k_inter(const float* __restrict__ x,
                                               const float* __restrict__ pw,
                                               const float* __restrict__ pb,
                                               float* __restrict__ sAcc,
                                               double* __restrict__ ssqAcc) {
  __shared__ __align__(16) float Af[2][128 * ASTR];  // 2 x 10 KB
  __shared__ __align__(16) float Bf[2][16 * BSTR];   // 2 x 8.5 KB
  __shared__ float  scol[128];
  __shared__ double dred[256];

  const int tid  = threadIdx.x;
  const int lane = tid & 31, w = tid >> 5;
  const int wm = w & 3, wn = w >> 2;
  const int l15 = lane & 15, lh = lane >> 4;
  const int bm = blockIdx.x, bn = blockIdx.y;
  const size_t rowBase = (size_t)bm * 128;
  const int colBase = bn * 128;

  if (tid < 128) scol[tid] = 0.f;

  // per-thread DMA slots (32 B each side, 16 B-aligned on both sides)
  const int rA = tid >> 1, hA = tid & 1;
  const float* gA = x + (rowBase + rA) * (size_t)Dn + hA * 8;
  const int   lAo = rA * ASTR + hA * 8;
  const int kB = tid >> 4, cB = (tid & 15) * 8;
  const float* gB = pw + (size_t)kB * DP + colBase + cB;
  const int   lBo = kB * BSTR + cB;

  // issue async DMA for K-stage k0 into buffer `buf`
  auto issue = [&](int k0, int buf) {
    async_copy32B(gA + k0, &Af[buf][lAo]);
    async_copy32B(gB + (size_t)k0 * DP, &Bf[buf][lBo]);
  };

  const v8f vzero = {0.f, 0.f, 0.f, 0.f, 0.f, 0.f, 0.f, 0.f};
  v8f acc[2][4];
#pragma unroll
  for (int mi = 0; mi < 2; ++mi)
#pragma unroll
    for (int ni = 0; ni < 4; ++ni) acc[mi][ni] = vzero;

  issue(0, 0);
  const int KT = Dn / 16;                  // 64 stages
  for (int kt = 0; kt < KT; ++kt) {
    const int cur = kt & 1;
    wait_async0();                         // stage kt landed in LDS
    __syncthreads();                       // visible to all waves
    if (kt + 1 < KT) issue((kt + 1) * 16, cur ^ 1);  // prefetch next stage

    const float* Ac = Af[cur];
    const float* Bc = Bf[cur];
#pragma unroll
    for (int ks = 0; ks < 4; ++ks) {
      v2f a2[2], b2[4];
#pragma unroll
      for (int mi = 0; mi < 2; ++mi)
        a2[mi] = *(const v2f*)&Ac[(wm * 32 + mi * 16 + l15) * ASTR + ks * 4 + 2 * lh];
#pragma unroll
      for (int ni = 0; ni < 4; ++ni) {
        int n = wn * 64 + ni * 16 + l15;
        int k = ks * 4 + 2 * lh;
        v2f b;
        b[0] = Bc[(k + 0) * BSTR + n];
        b[1] = Bc[(k + 1) * BSTR + n];
        b2[ni] = b;
      }
#pragma unroll
      for (int mi = 0; mi < 2; ++mi)
#pragma unroll
        for (int ni = 0; ni < 4; ++ni)
          acc[mi][ni] = __builtin_amdgcn_wmma_f32_16x16x4_f32(
              false, a2[mi], false, b2[ni], (short)0, acc[mi][ni], false, false);
    }
    __syncthreads();                       // done reading buf `cur`
  }

  // epilogue: xp = acc + pb; accumulate column sums (for ||Σxp||²) and Σxp²
  double dsq = 0.0;
#pragma unroll
  for (int ni = 0; ni < 4; ++ni) {
    int colLocal = wn * 64 + ni * 16 + l15;
    float bias = pb[colBase + colLocal];
    float csum = 0.f;
#pragma unroll
    for (int mi = 0; mi < 2; ++mi)
#pragma unroll
      for (int v = 0; v < 8; ++v) {
        float xp = acc[mi][ni][v] + bias;
        dsq += (double)xp * (double)xp;
        csum += xp;
      }
    atomicAdd(&scol[colLocal], csum);
  }
  dred[tid] = dsq;
  __syncthreads();

  int batch = bm >> 4;
  if (tid < 128) atomicAdd(&sAcc[batch * DP + colBase + tid], scol[tid]);
  for (int s = 128; s > 0; s >>= 1) {
    if (tid < s) dred[tid] += dred[tid + s];
    __syncthreads();
  }
  if (tid == 0) atomicAdd(&ssqAcc[batch], dred[0]);
}

// ---------------- global-energy MLP on mean-pooled g ----------------
// grid (16, 4): block = (batch b, 256-wide f-chunk)
__global__ __launch_bounds__(256) void k_globalmlp(const float* __restrict__ g,
                                                   const float* __restrict__ gw1,
                                                   const float* __restrict__ gb1,
                                                   const float* __restrict__ gw2,
                                                   float* __restrict__ gE) {
  __shared__ float gl[Dn];
  __shared__ float red[256];
  const int tid = threadIdx.x;
  const int b = blockIdx.y;
  const int f = blockIdx.x * 256 + tid;

#pragma unroll
  for (int j = 0; j < 4; ++j) gl[tid + j * 256] = g[b * Dn + tid + j * 256];
  __syncthreads();

  float acc = gb1[f];
  for (int d = 0; d < Dn; ++d) {
    __builtin_prefetch(&gw1[(size_t)(d + 16) * DFF + f], 0, 0);
    acc += gl[d] * gw1[(size_t)d * DFF + f];
  }
  red[tid] = gelu_exact(acc) * gw2[f];
  __syncthreads();
  for (int s = 128; s > 0; s >>= 1) {
    if (tid < s) red[tid] += red[tid + s];
    __syncthreads();
  }
  if (tid == 0) atomicAdd(&gE[b], red[0]);
}

// ---------------- finalize: out[b] = local + ½(||s||² − Σ||xp||²) + global ----------------
__global__ __launch_bounds__(256) void k_finalize(const float* __restrict__ sAcc,
                                                  const double* __restrict__ ssqAcc,
                                                  const float* __restrict__ localAcc,
                                                  const float* __restrict__ gE,
                                                  const float* __restrict__ lb2,
                                                  const float* __restrict__ gb2,
                                                  float* __restrict__ out) {
  __shared__ double dr[256];
  __shared__ double resS[Bn];
  const int tid = threadIdx.x;
  for (int b = 0; b < Bn; ++b) {
    float sv = sAcc[b * DP + tid];           // DP == 256 == blockDim
    dr[tid] = (double)sv * (double)sv;
    __syncthreads();
    for (int s = 128; s > 0; s >>= 1) {
      if (tid < s) dr[tid] += dr[tid + s];
      __syncthreads();
    }
    if (tid == 0) resS[b] = dr[0];
    __syncthreads();
  }
  if (tid < Bn) {
    double inter = 0.5 * (resS[tid] - ssqAcc[tid]);
    out[tid] = (float)((double)localAcc[tid] + (double)Nn * (double)lb2[0] +
                       inter + (double)gE[tid] + (double)gb2[0]);
  }
}

extern "C" void kernel_launch(void* const* d_in, const int* in_sizes, int n_in,
                              void* d_out, int out_size, void* d_ws, size_t ws_size,
                              hipStream_t stream) {
  (void)in_sizes; (void)n_in; (void)out_size; (void)ws_size;
  const float* x   = (const float*)d_in[0];
  const float* lw1 = (const float*)d_in[1];
  const float* lb1 = (const float*)d_in[2];
  const float* lw2 = (const float*)d_in[3];
  const float* lb2 = (const float*)d_in[4];
  const float* pw  = (const float*)d_in[5];
  const float* pb  = (const float*)d_in[6];
  const float* gw1 = (const float*)d_in[7];
  const float* gb1 = (const float*)d_in[8];
  const float* gw2 = (const float*)d_in[9];
  const float* gb2 = (const float*)d_in[10];

  char* ws = (char*)d_ws;
  float*  ws_g     = (float*)(ws + OFF_G);
  float*  ws_s     = (float*)(ws + OFF_S);
  float*  ws_local = (float*)(ws + OFF_LOCAL);
  float*  ws_gE    = (float*)(ws + OFF_GE);
  double* ws_ssq   = (double*)(ws + OFF_SSQ);

  k_zero<<<dim3((ZERO_DW + 255) / 256), dim3(256), 0, stream>>>(
      (unsigned int*)(ws + OFF_S), ZERO_DW);
  k_colmean<<<dim3((Bn * Dn) / 256), dim3(256), 0, stream>>>(x, ws_g);
  k_localmlp<<<dim3((Bn * Nn) / 128, DFF / 128), dim3(256), 0, stream>>>(
      x, lw1, lb1, lw2, ws_local);
  k_inter<<<dim3((Bn * Nn) / 128, DP / 128), dim3(256), 0, stream>>>(
      x, pw, pb, ws_s, ws_ssq);
  k_globalmlp<<<dim3(DFF / 256, Bn), dim3(256), 0, stream>>>(
      ws_g, gw1, gb1, gw2, ws_gE);
  k_finalize<<<dim3(1), dim3(256), 0, stream>>>(
      ws_s, ws_ssq, ws_local, ws_gE, lb2, gb2, (float*)d_out);
}
